// DQN_57904749085018
// MI455X (gfx1250) — compile-verified, matrix operand order checked by reading, not run on hardware
//
#include <hip/hip_runtime.h>

// Problem constants (reference: S=100, L=13, B=8192, FEAT=4*S+7)
constexpr int NS   = 100;
constexpr int NL   = 13;
constexpr int NB   = 8192;
constexpr int FEAT = 407;
#define EPS_F 1e-5f

typedef __attribute__((ext_vector_type(16))) _Float16 v16h;
typedef __attribute__((ext_vector_type(8)))  float    v8f;

// tanh(y) = 1 - 2/(exp(2y)+1), exp(2y) = exp2(y * 2*log2(e)).
// Uses full-rate v_exp_f32 / v_rcp_f32 transcendentals (~1ulp), saturates
// correctly for |y| large.
__device__ __forceinline__ float fast_tanh(float y) {
    float e = __builtin_amdgcn_exp2f(y * 2.8853900817779268f); // 2/ln(2)
    float r = __builtin_amdgcn_rcpf(e + 1.0f);
    return fmaf(-2.0f, r, 1.0f);
}

// ---------------------------------------------------------------------------
// Stage 1: gather + normalize + tiny MLP (13->3->2, tanh).
// One thread per (b, s). Writes:
//   aT [NS][NB]  (transposed so stage-2 reads a-rows coalesced)
//   sbM[NB][NS]
// All parameter tables are tiny (<100 KB) and L2-resident; x row (1.6 KB) is
// shared by the 32 lanes of a wave (s fastest in tid), so reads are cache-hot.
// ---------------------------------------------------------------------------
__global__ void dqn_stage1(const float* __restrict__ x,
                           const float* __restrict__ gamma,
                           const float* __restrict__ beta,
                           const float* __restrict__ mean,
                           const float* __restrict__ var,
                           const float* __restrict__ W1,
                           const float* __restrict__ b1,
                           const float* __restrict__ W2,
                           const float* __restrict__ b2,
                           const int*   __restrict__ idx,
                           float* __restrict__ aT,
                           float* __restrict__ sbM) {
    int tid = blockIdx.x * blockDim.x + threadIdx.x;
    if (tid >= NB * NS) return;
    int s = tid % NS;
    int b = tid / NS;

    const float* xr = x + (size_t)b * FEAT;

    float h[NL];
#pragma unroll
    for (int l = 0; l < NL; ++l) {
        int p = s * NL + l;
        float g   = xr[idx[p]];
        float inv = rsqrtf(var[p] + EPS_F);
        h[l] = fmaf((g - mean[p]) * inv, gamma[p], beta[p]);
    }

    float h1[3];
#pragma unroll
    for (int o = 0; o < 3; ++o) {
        const float* w = W1 + (size_t)(s * 3 + o) * NL;
        float acc = b1[s * 3 + o];
#pragma unroll
        for (int l = 0; l < NL; ++l) acc = fmaf(w[l], h[l], acc);
        h1[o] = fast_tanh(acc);
    }

    float h2[2];
#pragma unroll
    for (int o = 0; o < 2; ++o) {
        const float* w = W2 + (size_t)(s * 2 + o) * 3;
        float acc = b2[s * 2 + o];
#pragma unroll
        for (int k = 0; k < 3; ++k) acc = fmaf(w[k], h1[k], acc);
        h2[o] = fast_tanh(acc);
    }

    aT [(size_t)s * NB + b] = h2[0];   // a,  transposed (i-major)
    sbM[(size_t)b * NS + s] = h2[1];   // sb, b-major
}

// ---------------------------------------------------------------------------
// Stage 2: out[b,i,j] = tanh(a[b,i]*W3[i,j,0] + sb[b,j]*W3[i,j,1] + b3[i,j])
//
// One wave (32 threads) per block => EXEC all-1s for WMMA.
// Tile: 16 b-rows x 16 j-cols at fixed i, as D = A x B + C with
//   A[m,0]   = a[bt+m, i]            (f16, K=0 only, rest zero)
//   B[0,n]   = W3[i, jt+n, 0]        (f16, K=0 only, rest zero)
//   C[m,n]   = sb[bt+m, jt+n]*W3[i,jt+n,1] + b3[i,jt+n]   (f32 accumulator)
// then tanh(D) -> non-temporal stores (write-once 328 MB; keep L2 for tables).
//
// VGPR layouts (CDNA5 ISA):
//   16-bit A 16x32: lane m<16 holds row m, h[0]=K0  -> A[0]=a (lanes 0-15)
//                   lanes 16-31 hold K=8..          -> must be 0
//   16-bit B 32x16: lane n<16 holds col n, h[0]=K0  -> B[0]=w3a (lanes 0-15)
//                   lanes 16-31 hold K=8..         -> must be 0
//   f32 C/D 16x16 : VGPR r -> row M=r (lanes 0-15) / M=8+r (lanes 16-31)
// ---------------------------------------------------------------------------
__global__ void __launch_bounds__(32)
dqn_stage2(const float* __restrict__ aT,    // [NS][NB]
           const float* __restrict__ sbM,   // [NB][NS]
           const float* __restrict__ W3,    // [NS][NS][2]
           const float* __restrict__ b3,    // [NS][NS]
           float* __restrict__ out) {       // [NB][NS*NS]
    const int lane = threadIdx.x;       // 0..31
    const int n    = lane & 15;         // column within tile
    const int hi   = lane >> 4;         // row-half select (M = r + 8*hi)
    const int bt   = blockIdx.x * 16;
    const int jt   = blockIdx.y * 16;
    const int i0   = blockIdx.z * 25;

    const int  j     = jt + n;
    const bool valid = (j < NS);
    const int  jv    = valid ? j : (NS - 1);   // clamp for loads, mask stores

    // sb for this lane's column, 8 b-rows; invariant over the i loop.
    float sbv[8];
#pragma unroll
    for (int r = 0; r < 8; ++r)
        sbv[r] = sbM[(size_t)(bt + hi * 8 + r) * NS + jv];

    const _Float16 hz = (_Float16)0.0f;

    for (int i = i0; i < i0 + 25; ++i) {
        // W3[i,j,0..1] as one 8-byte load (contiguous pair), b3[i,j].
        const float2 w3  = *reinterpret_cast<const float2*>(
                               W3 + ((size_t)i * NS + jv) * 2);
        const float  b3v = b3[(size_t)i * NS + jv];
        // a-row: coalesced 64B segment (lanes 16-31 replicate, then zeroed).
        const float  av  = aT[(size_t)i * NB + bt + n];

        v16h A = {};
        v16h B = {};
        A[0] = hi ? hz : (_Float16)av;     // v_cndmask, no divergence
        B[0] = hi ? hz : (_Float16)w3.x;

        v8f C;
#pragma unroll
        for (int r = 0; r < 8; ++r)
            C[r] = fmaf(sbv[r], w3.y, b3v);

        v8f D = __builtin_amdgcn_wmma_f32_16x16x32_f16(
            false, A, false, B, (short)0, C, false, false);

        if (valid) {
            float* op = out + (size_t)(bt + hi * 8) * (NS * NS)
                            + (size_t)i * NS + j;
#pragma unroll
            for (int r = 0; r < 8; ++r)
                __builtin_nontemporal_store(fast_tanh(D[r]),
                                            op + (size_t)r * (NS * NS));
        }
    }
}

// ---------------------------------------------------------------------------
extern "C" void kernel_launch(void* const* d_in, const int* in_sizes, int n_in,
                              void* d_out, int out_size, void* d_ws, size_t ws_size,
                              hipStream_t stream) {
    const float* x     = (const float*)d_in[0];
    const float* gamma = (const float*)d_in[1];
    const float* beta  = (const float*)d_in[2];
    const float* mean  = (const float*)d_in[3];
    const float* var   = (const float*)d_in[4];
    const float* W1    = (const float*)d_in[5];
    const float* b1    = (const float*)d_in[6];
    const float* W2    = (const float*)d_in[7];
    const float* b2    = (const float*)d_in[8];
    const float* W3    = (const float*)d_in[9];
    const float* b3    = (const float*)d_in[10];
    const int*   idx   = (const int*)d_in[11];

    float* aT  = (float*)d_ws;                     // NS*NB floats (3.3 MB)
    float* sbM = aT + (size_t)NS * NB;             // NB*NS floats (3.3 MB)
    float* out = (float*)d_out;

    // Stage 1: one thread per (b, s)
    dim3 g1((NB * NS + 255) / 256);
    dqn_stage1<<<g1, 256, 0, stream>>>(x, gamma, beta, mean, var,
                                       W1, b1, W2, b2, idx, aT, sbM);

    // Stage 2: (b-tile, j-tile, i-chunk) x 1 wave; 512*7*4 = 14336 waves
    dim3 g2(NB / 16, (NS + 15) / 16, 4);
    dqn_stage2<<<g2, 32, 0, stream>>>(aT, sbM, W3, b3, out);
}
